// MultiHeadAttention_37881611550863
// MI455X (gfx1250) — matrix operand order repaired and checked by reading, not compiled, not run on hardware
//
#include <hip/hip_runtime.h>
#include <hip/hip_bf16.h>

// ---------------------------------------------------------------------------
// MultiHeadAttention for MI455X (gfx1250): bf16 WMMA pipeline.
// B=2, S=2048, D=1024, H=16, DK=DV=64.
// ---------------------------------------------------------------------------

constexpr int cB  = 2;
constexpr int cS  = 2048;
constexpr int cD  = 1024;
constexpr int cH  = 16;
constexpr int cDK = 64;
constexpr int cDV = 64;
constexpr int cHD = cH * cDV;     // 1024
constexpr int cBS = cB * cS;      // 4096
constexpr int cSCP = 2052;        // padded LDS score row stride (floats)

typedef __bf16 bf16_t;
typedef bf16_t v16bf __attribute__((ext_vector_type(16)));
typedef float  v8f   __attribute__((ext_vector_type(8)));
typedef float  v4f   __attribute__((ext_vector_type(4)));
typedef unsigned int v4u __attribute__((ext_vector_type(4)));

union BfVec {
  v16bf v;
  v4u q[2];
  unsigned short s[16];
};

__device__ __forceinline__ unsigned short f2bfbits(float f) {
  unsigned u = __float_as_uint(f);
  return (unsigned short)((u + 0x7FFFu + ((u >> 16) & 1u)) >> 16);  // RNE
}

__device__ __forceinline__ v8f vzero8() {
  v8f z;
#pragma unroll
  for (int i = 0; i < 8; ++i) z[i] = 0.0f;
  return z;
}

__device__ __forceinline__ v8f wmma_bf16(v16bf a, v16bf b, v8f c) {
  // D = A(16x32 bf16) * B(32x16 bf16) + C(16x16 f32)
  return __builtin_amdgcn_wmma_f32_16x16x32_bf16(
      /*neg_a=*/false, a, /*neg_b=*/false, b,
      /*c_mod=*/(short)0, c, /*reuse_a=*/false, /*reuse_b=*/false);
}

// ---------------------------------------------------------------------------
// Conversion / layout kernels
// ---------------------------------------------------------------------------
__global__ void cvt_f32_bf16(const float* __restrict__ in,
                             unsigned short* __restrict__ out, int n) {
  int i = blockIdx.x * blockDim.x + threadIdx.x;
  if (i < n) out[i] = f2bfbits(in[i]);
}

// in [H, D, DK] -> out [H, DK, D] (bf16)
__global__ void cvt_T_head(const float* __restrict__ in,
                           unsigned short* __restrict__ out) {
  int i = blockIdx.x * blockDim.x + threadIdx.x;
  int h = i / (cD * cDK);
  int r = i % (cD * cDK);
  int d = r / cDK;
  int n = r % cDK;
  out[((size_t)h * cDK + n) * cD + d] = f2bfbits(in[i]);
}

// in [HD, D] -> out [D, HD] (bf16)
__global__ void cvt_T2(const float* __restrict__ in,
                       unsigned short* __restrict__ out) {
  int i = blockIdx.x * blockDim.x + threadIdx.x;
  int k = i / cD;
  int n = i % cD;
  out[(size_t)n * cHD + k] = f2bfbits(in[i]);
}

// ---------------------------------------------------------------------------
// q/k/v projection: X[BS,D](bf16) x W^T[H,DK,D](bf16) + bias -> per-head tiles.
// q,k stored [B,H,S,DK]; v stored transposed [B,H,DV,S] for the ctx GEMM.
// Block = 128 threads (4 waves); wave computes a 16x64 output tile.
// ---------------------------------------------------------------------------
__global__ void proj_qkv_kernel(
    const unsigned short* __restrict__ Xq, const unsigned short* __restrict__ Xk,
    const unsigned short* __restrict__ Xv,
    const unsigned short* __restrict__ WqT, const unsigned short* __restrict__ WkT,
    const unsigned short* __restrict__ WvT,
    const float* __restrict__ bq, const float* __restrict__ bk,
    const float* __restrict__ bv,
    unsigned short* __restrict__ qO, unsigned short* __restrict__ kO,
    unsigned short* __restrict__ vO) {
  const int z = blockIdx.z;
  const unsigned short* X;
  const unsigned short* WT;
  const float* bias;
  unsigned short* out;
  if (z == 0)      { X = Xq; WT = WqT; bias = bq; out = qO; }
  else if (z == 1) { X = Xk; WT = WkT; bias = bk; out = kO; }
  else             { X = Xv; WT = WvT; bias = bv; out = vO; }

  const int h    = blockIdx.y;
  const int wave = threadIdx.x >> 5;
  const int lane = threadIdx.x & 31;
  const int l    = lane & 15;
  const int half = lane >> 4;
  const int m0   = blockIdx.x * 64 + wave * 16;

  const unsigned short* arow = X + (size_t)(m0 + l) * cD;
  const unsigned short* wth  = WT + (size_t)h * cDK * cD;

  v8f acc[4];
#pragma unroll
  for (int i = 0; i < 4; ++i) acc[i] = vzero8();

  for (int kk = 0; kk < cD / 32; ++kk) {
    const int k0 = kk * 32;
    __builtin_prefetch(arow + k0 + 128, 0, 0);
    BfVec a;
    a.q[0] = *(const v4u*)(arow + k0 + half * 16);
    a.q[1] = *(const v4u*)(arow + k0 + half * 16 + 8);
#pragma unroll
    for (int nt = 0; nt < 4; ++nt) {
      const unsigned short* wrow =
          wth + (size_t)(nt * 16 + l) * cD + k0 + half * 16;
      BfVec bm;
      bm.q[0] = *(const v4u*)(wrow);
      bm.q[1] = *(const v4u*)(wrow + 8);
      acc[nt] = wmma_bf16(a.v, bm.v, acc[nt]);
    }
  }

  const int rbase = half * 8;
  const int bb    = m0 >> 11;            // m0 / S
  const int sbase = m0 & (cS - 1);       // m0 % S

  if (z < 2) {
    unsigned short* obase = out + ((size_t)(bb * cH + h) * cS) * cDK;
#pragma unroll
    for (int nt = 0; nt < 4; ++nt) {
      const int n = nt * 16 + l;
      const float bn = bias[h * cDK + n];
#pragma unroll
      for (int r = 0; r < 8; ++r) {
        const int s = sbase + rbase + r;
        obase[(size_t)s * cDK + n] = f2bfbits(acc[nt][r] + bn);
      }
    }
  } else {
#pragma unroll
    for (int nt = 0; nt < 4; ++nt) {
      const int n = nt * 16 + l;
      const float bn = bias[h * cDV + n];
      unsigned short tmp[8];
#pragma unroll
      for (int r = 0; r < 8; ++r) tmp[r] = f2bfbits(acc[nt][r] + bn);
      v4u pk;
      pk.x = (unsigned)tmp[0] | ((unsigned)tmp[1] << 16);
      pk.y = (unsigned)tmp[2] | ((unsigned)tmp[3] << 16);
      pk.z = (unsigned)tmp[4] | ((unsigned)tmp[5] << 16);
      pk.w = (unsigned)tmp[6] | ((unsigned)tmp[7] << 16);
      *(v4u*)(out + ((size_t)(bb * cH + h) * cDV + n) * cS + sbase + rbase) = pk;
    }
  }
}

// ---------------------------------------------------------------------------
// Fused attention: one block (4 waves, 128 threads) cooperates on a single
// 16-row q-tile of one (b,h). LDS = 16 x 2052 f32 = 131,328 B -> 2 blocks
// (8 waves) resident per WGP (320 KB LDS).
//   Phase 1: scores = q k^T via WMMA -> shared LDS rows (waves split s-tiles)
//   Phase 2: row softmax (waves split rows), NT-store attn f32, probs -> LDS
//   Phase 3: ctx = attn x v via WMMA, waves split K; LDS tree-reduce partials.
// ---------------------------------------------------------------------------
__global__ void attn_kernel(const unsigned short* __restrict__ qp,
                            const unsigned short* __restrict__ kp,
                            const unsigned short* __restrict__ vp,
                            unsigned short* __restrict__ ctxp,
                            float* __restrict__ attnO) {
  extern __shared__ float smem[];  // 16 * 2052 floats
  const int b    = blockIdx.z;
  const int h    = blockIdx.y;
  const int wave = threadIdx.x >> 5;
  const int lane = threadIdx.x & 31;
  const int l    = lane & 15;
  const int half = lane >> 4;
  const int q0   = blockIdx.x * 16;

#define SCr(r) (smem + (r) * cSCP)

  const size_t bhS = ((size_t)b * cH + h) * cS;

  // Hoisted A operands: q tile 16x64 (two 16x32 chunks), loaded by each wave.
  const unsigned short* qrow = qp + (bhS + q0 + l) * cDK;
  BfVec a0, a1;
  a0.q[0] = *(const v4u*)(qrow + half * 16);
  a0.q[1] = *(const v4u*)(qrow + half * 16 + 8);
  a1.q[0] = *(const v4u*)(qrow + 32 + half * 16);
  a1.q[1] = *(const v4u*)(qrow + 40 + half * 16);

  // ---- Phase 1: scores (waves split the 128 s-tiles, disjoint columns) ----
  for (int st = wave; st < cS / 16; st += 4) {
    const int s0 = st * 16;
    const unsigned short* krow = kp + (bhS + s0 + l) * cDK + half * 16;
    BfVec b0, b1;
    b0.q[0] = *(const v4u*)(krow);
    b0.q[1] = *(const v4u*)(krow + 8);
    b1.q[0] = *(const v4u*)(krow + 32);
    b1.q[1] = *(const v4u*)(krow + 40);
    v8f accv = wmma_bf16(a0.v, b0.v, vzero8());
    accv = wmma_bf16(a1.v, b1.v, accv);
#pragma unroll
    for (int r = 0; r < 8; ++r) SCr(half * 8 + r)[s0 + l] = accv[r];
  }
  __syncthreads();

  // ---- Phase 2: softmax (waves split the 16 rows, 4 each) ----
  const float scale = 0.125f;  // 1/sqrt(DK)
#pragma unroll 1
  for (int rr = 0; rr < 4; ++rr) {
    const int r = wave * 4 + rr;
    float* row = SCr(r);
    float mx = -3.0e38f;
    for (int i = lane; i < cS; i += 32) mx = fmaxf(mx, row[i]);
#pragma unroll
    for (int off = 16; off > 0; off >>= 1) mx = fmaxf(mx, __shfl_xor(mx, off, 32));
    mx *= scale;
    float sum = 0.0f;
    for (int i = lane; i < cS; i += 32) {
      float e = __expf(row[i] * scale - mx);
      row[i] = e;
      sum += e;
    }
#pragma unroll
    for (int off = 16; off > 0; off >>= 1) sum += __shfl_xor(sum, off, 32);
    const float inv = 1.0f / sum;
    float* orow = attnO + ((size_t)(b * cS + q0 + r)) * (cH * cS) + (size_t)h * cS;
    for (int i = lane; i < cS; i += 32) {
      float p = row[i] * inv;
      row[i] = p;
      __builtin_nontemporal_store(p, orow + i);  // write-once 536MB stream
    }
  }
  __syncthreads();

  // ---- Phase 3: ctx = attn x v; waves split the K (=s) dimension ----
  v8f acc[4];
#pragma unroll
  for (int i = 0; i < 4; ++i) acc[i] = vzero8();

  const float* arow = SCr(l);
  const int kt0 = wave * 16;
  for (int kt = kt0; kt < kt0 + 16; ++kt) {
    const int sb = kt * 32 + half * 16;
    BfVec a;
#pragma unroll
    for (int i = 0; i < 8; ++i) a.s[i] = f2bfbits(arow[sb + i]);
#pragma unroll
    for (int i = 0; i < 8; ++i) a.s[8 + i] = f2bfbits(arow[sb + 8 + i]);
#pragma unroll
    for (int nt = 0; nt < 4; ++nt) {
      const unsigned short* vrow =
          vp + ((size_t)(b * cH + h) * cDV + nt * 16 + l) * cS + kt * 32 + half * 16;
      BfVec bm;
      bm.q[0] = *(const v4u*)(vrow);
      bm.q[1] = *(const v4u*)(vrow + 8);
      acc[nt] = wmma_bf16(a.v, bm.v, acc[nt]);
    }
  }
  __syncthreads();  // all reads of score LDS done; safe to reuse for partials

  // Each wave deposits its 16x64 f32 partial at smem[wave*1024].
  float* part = smem + wave * 1024;
#pragma unroll
  for (int nt = 0; nt < 4; ++nt)
#pragma unroll
    for (int r = 0; r < 8; ++r)
      part[(half * 8 + r) * 64 + nt * 16 + l] = acc[nt][r];
  __syncthreads();

  // Tree-reduce 4 partials; each thread owns 8 consecutive ctx elements.
  {
    const int e0  = threadIdx.x * 8;   // 0..1016 over 16x64 tile
    const int row = e0 >> 6;
    const int c0  = e0 & 63;
    v4f sa = *(const v4f*)(smem + e0) + *(const v4f*)(smem + 1024 + e0) +
             *(const v4f*)(smem + 2048 + e0) + *(const v4f*)(smem + 3072 + e0);
    v4f sb = *(const v4f*)(smem + e0 + 4) + *(const v4f*)(smem + 1024 + e0 + 4) +
             *(const v4f*)(smem + 2048 + e0 + 4) + *(const v4f*)(smem + 3072 + e0 + 4);
    unsigned short tmp[8];
#pragma unroll
    for (int j = 0; j < 4; ++j) { tmp[j] = f2bfbits(sa[j]); tmp[4 + j] = f2bfbits(sb[j]); }
    v4u pk;
    pk.x = (unsigned)tmp[0] | ((unsigned)tmp[1] << 16);
    pk.y = (unsigned)tmp[2] | ((unsigned)tmp[3] << 16);
    pk.z = (unsigned)tmp[4] | ((unsigned)tmp[5] << 16);
    pk.w = (unsigned)tmp[6] | ((unsigned)tmp[7] << 16);
    const int qg = q0 + row;
    *(v4u*)(ctxp + ((size_t)(b * cS + qg)) * cHD + h * cDV + c0) = pk;
  }
#undef SCr
}

// ---------------------------------------------------------------------------
// Output projection: ctx[BS,HD](bf16) x Wm^T[D,HD](bf16) + bm -> f32 out.
// ---------------------------------------------------------------------------
__global__ void outproj_kernel(const unsigned short* __restrict__ ctxp,
                               const unsigned short* __restrict__ WmT,
                               const float* __restrict__ bm,
                               float* __restrict__ mh) {
  const int wave = threadIdx.x >> 5;
  const int lane = threadIdx.x & 31;
  const int l    = lane & 15;
  const int half = lane >> 4;
  const int m0   = blockIdx.x * 64 + wave * 16;
  const int n0   = blockIdx.y * 64;

  const unsigned short* arow = ctxp + (size_t)(m0 + l) * cHD;

  v8f acc[4];
#pragma unroll
  for (int i = 0; i < 4; ++i) acc[i] = vzero8();

  for (int kk = 0; kk < cHD / 32; ++kk) {
    const int k0 = kk * 32;
    __builtin_prefetch(arow + k0 + 128, 0, 0);
    BfVec a;
    a.q[0] = *(const v4u*)(arow + k0 + half * 16);
    a.q[1] = *(const v4u*)(arow + k0 + half * 16 + 8);
#pragma unroll
    for (int nt = 0; nt < 4; ++nt) {
      const unsigned short* wrow =
          WmT + (size_t)(n0 + nt * 16 + l) * cHD + k0 + half * 16;
      BfVec bm2;
      bm2.q[0] = *(const v4u*)(wrow);
      bm2.q[1] = *(const v4u*)(wrow + 8);
      acc[nt] = wmma_bf16(a.v, bm2.v, acc[nt]);
    }
  }

#pragma unroll
  for (int nt = 0; nt < 4; ++nt) {
    const int n = n0 + nt * 16 + l;
    const float bn = bm[n];
#pragma unroll
    for (int r = 0; r < 8; ++r) {
      const int mg = m0 + half * 8 + r;
      mh[(size_t)mg * cD + n] = acc[nt][r] + bn;
    }
  }
}

// ---------------------------------------------------------------------------
extern "C" void kernel_launch(void* const* d_in, const int* in_sizes, int n_in,
                              void* d_out, int out_size, void* d_ws,
                              size_t ws_size, hipStream_t stream) {
  (void)in_sizes; (void)n_in; (void)out_size; (void)ws_size;

  const float* Q  = (const float*)d_in[0];
  const float* K  = (const float*)d_in[1];
  const float* V  = (const float*)d_in[2];
  const float* Wq = (const float*)d_in[3];
  const float* bq = (const float*)d_in[4];
  const float* Wk = (const float*)d_in[5];
  const float* bk = (const float*)d_in[6];
  const float* Wv = (const float*)d_in[7];
  const float* bv = (const float*)d_in[8];
  const float* Wm = (const float*)d_in[9];
  const float* bm = (const float*)d_in[10];

  char* ws = (char*)d_ws;
  const size_t MB = 1u << 20;
  unsigned short* Xq   = (unsigned short*)(ws + 0 * MB);    // 8 MB each
  unsigned short* Xk   = (unsigned short*)(ws + 8 * MB);
  unsigned short* Xv   = (unsigned short*)(ws + 16 * MB);
  unsigned short* WqT  = (unsigned short*)(ws + 24 * MB);   // 2 MB each
  unsigned short* WkT  = (unsigned short*)(ws + 26 * MB);
  unsigned short* WvT  = (unsigned short*)(ws + 28 * MB);
  unsigned short* WmT  = (unsigned short*)(ws + 30 * MB);
  unsigned short* qp   = (unsigned short*)(ws + 32 * MB);   // 8 MB each
  unsigned short* kp   = (unsigned short*)(ws + 40 * MB);
  unsigned short* vp   = (unsigned short*)(ws + 48 * MB);   // [B,H,DV,S]
  unsigned short* ctxp = (unsigned short*)(ws + 56 * MB);   // [BS,HD]

  float* mh    = (float*)d_out;
  float* attnO = mh + (size_t)cBS * cD;

  const int nX = cBS * cD;               // 4,194,304
  cvt_f32_bf16<<<nX / 256, 256, 0, stream>>>(Q, Xq, nX);
  cvt_f32_bf16<<<nX / 256, 256, 0, stream>>>(K, Xk, nX);
  cvt_f32_bf16<<<nX / 256, 256, 0, stream>>>(V, Xv, nX);
  const int nW = cH * cD * cDK;          // 1,048,576
  cvt_T_head<<<nW / 256, 256, 0, stream>>>(Wq, WqT);
  cvt_T_head<<<nW / 256, 256, 0, stream>>>(Wk, WkT);
  cvt_T_head<<<nW / 256, 256, 0, stream>>>(Wv, WvT);
  cvt_T2<<<(cHD * cD) / 256, 256, 0, stream>>>(Wm, WmT);

  proj_qkv_kernel<<<dim3(cBS / 64, cH, 3), 128, 0, stream>>>(
      Xq, Xk, Xv, WqT, WkT, WvT, bq, bk, bv, qp, kp, vp);

  attn_kernel<<<dim3(cS / 16, cH, cB), 128, 16 * cSCP * sizeof(float),
                stream>>>(qp, kp, vp, ctxp, attnO);

  outproj_kernel<<<dim3(cBS / 64, cD / 64), 128, 0, stream>>>(ctxp, WmT, bm, mh);
}
